// LePEAttention_45810121179554
// MI455X (gfx1250) — matrix operand order repaired and checked by reading, not compiled, not run on hardware
//
#include <hip/hip_runtime.h>

typedef _Float16 v16h __attribute__((ext_vector_type(16)));
typedef _Float16 v8h  __attribute__((ext_vector_type(8)));
typedef float    v8f  __attribute__((ext_vector_type(8)));

#define Bn    8
#define HWdim 64
#define Cc    256
#define HEADS 8
#define HD    32            // head dim
#define H_SP  64
#define W_SP  8
#define Ltot  (HWdim*HWdim) // 4096
#define Twin  (H_SP*W_SP)   // 512 tokens per window
#define NWIN  (Bn*(HWdim/W_SP)) // 64 windows
#define BM    64            // q rows per workgroup

// Padded LDS row pitches (avoid 64-bank conflicts; keep 16B alignment)
#define SLD 516             // S  row pitch in f32  (bank stride 4)
#define QLD 40              // Qs row pitch in f16  (bank stride 20)
#define KLD 40              // Ks row pitch in f16  (bank stride 20)
#define VLD 520             // Vt row pitch in f16  (bank stride 4)

#define S_BYTES  (BM*SLD*4)        // 132096
#define Q_BYTES  (BM*QLD*2)        // 5120
#define K_BYTES  (Twin*KLD*2)      // 40960
#define V_BYTES  (HD*VLD*2)        // 33280
#define SMEM_BYTES (S_BYTES+Q_BYTES+K_BYTES+V_BYTES)  // 211456

// ---------------- Kernel 1: depthwise 3x3 LePE conv + bias -> out region ----
__global__ __launch_bounds__(256)
void lepe_kernel(const float* __restrict__ qkv,
                 const float* __restrict__ w,
                 const float* __restrict__ bias,
                 float* __restrict__ out)
{
    int idx = blockIdx.x * 256 + threadIdx.x;     // over B*L*C (exact multiple)
    int c  = idx % Cc;
    int bl = idx / Cc;
    int l  = bl % Ltot;
    int b  = bl / Ltot;
    int y   = l >> 6;       // row in 64x64 image == row in window image
    int col = l & 63;
    int xw  = col >> 3;     // which width-window
    int x   = col & 7;      // column inside window (0..7)
    const float* v = qkv + (size_t)2 * Bn * Ltot * Cc;
    float acc = bias[c];
    #pragma unroll
    for (int dy = -1; dy <= 1; ++dy) {
        int yy = y + dy;
        if (yy < 0 || yy >= H_SP) continue;
        #pragma unroll
        for (int dx = -1; dx <= 1; ++dx) {
            int xx = x + dx;
            if (xx < 0 || xx >= W_SP) continue;   // zero-pad at window edge
            int ll = yy * HWdim + xw * W_SP + xx;
            acc += w[c * 9 + (dy + 1) * 3 + (dx + 1)]
                 * v[((size_t)b * Ltot + ll) * Cc + c];
        }
    }
    out[idx] = acc;
}

// ---------------- Kernel 2: windowed attention with WMMA ----------
__global__ __launch_bounds__(256)
void attn_kernel(const float* __restrict__ qkv, float* __restrict__ out)
{
    extern __shared__ __align__(128) char smem[];
    float*    S  = (float*)smem;                                 // 64 x SLD f32
    _Float16* Qs = (_Float16*)(smem + S_BYTES);                  // 64 x QLD f16
    _Float16* Ks = (_Float16*)(smem + S_BYTES + Q_BYTES);        // 512 x KLD f16
    _Float16* Vt = (_Float16*)(smem + S_BYTES + Q_BYTES + K_BYTES); // 32 x VLD

    const int tid  = threadIdx.x;
    const int lane = tid & 31;
    const int wv   = tid >> 5;
    const int rb   = blockIdx.x;   // row block 0..7
    const int h    = blockIdx.y;   // head 0..7
    const int win  = blockIdx.z;   // window 0..63
    const int b    = win >> 3;
    const int xw   = win & 7;

    const float scale = 0.17677669529663687f;  // 32^-0.5

    const size_t qbase = (size_t)b * Ltot * Cc + (size_t)h * HD;
    const float* Q = qkv;
    const float* K = qkv + (size_t)Bn * Ltot * Cc;
    const float* V = qkv + (size_t)2 * Bn * Ltot * Cc;

    // ---- stage Q (scaled), K, V^T into LDS as f16 ----
    #pragma unroll
    for (int i = 0; i < 8; ++i) {              // 64*32 = 2048 elems
        int e = tid + 256 * i;
        int r = e >> 5, d = e & 31;
        int t = rb * BM + r;
        int l = (t >> 3) * HWdim + xw * W_SP + (t & 7);
        Qs[r * QLD + d] = (_Float16)(Q[qbase + (size_t)l * Cc + d] * scale);
    }
    for (int i = 0; i < 64; ++i) {             // 512*32 = 16384 elems
        int e = tid + 256 * i;
        int t = e >> 5, d = e & 31;
        int l = (t >> 3) * HWdim + xw * W_SP + (t & 7);
        size_t off = qbase + (size_t)l * Cc + d;
        Ks[t * KLD + d]  = (_Float16)K[off];
        Vt[d * VLD + t]  = (_Float16)V[off];
    }
    __syncthreads();

    // ---- S = Q K^T : 4x32 tiles of 16x16, 16 tiles per wave ----
    {
        int mt    = wv >> 1;
        int nhalf = wv & 1;
        // A operand (16x32 f16): lanes 0-15 -> K 0-7,16-23 ; lanes 16-31 -> K 8-15,24-31
        int m    = mt * 16 + (lane & 15);
        int koff = (lane >> 4) * 8;
        v8h alo = *(const v8h*)(Qs + m * QLD + koff);
        v8h ahi = *(const v8h*)(Qs + m * QLD + 16 + koff);
        v16h a;
        #pragma unroll
        for (int i = 0; i < 8; ++i) { a[i] = alo[i]; a[8 + i] = ahi[i]; }

        int n     = lane & 15;
        int kboff = (lane >> 4) * 8;    // B: lanes 0-15 -> K 0-7,16-23 ; lanes 16-31 -> K 8-15,24-31
        for (int nt = 0; nt < 16; ++nt) {
            int ntile = nhalf * 16 + nt;
            int krow  = ntile * 16 + n;           // K-matrix token row
            v8h blo = *(const v8h*)(Ks + krow * KLD + kboff);
            v8h bhi = *(const v8h*)(Ks + krow * KLD + kboff + 16);
            v16h bm;
            #pragma unroll
            for (int i = 0; i < 8; ++i) { bm[i] = blo[i]; bm[8 + i] = bhi[i]; }
            v8f s = {};
            s = __builtin_amdgcn_wmma_f32_16x16x32_f16(
                    false, a, false, bm, (short)0, s, false, false);
            int mbase = mt * 16 + ((lane >> 4) << 3);
            int colq  = ntile * 16 + n;
            #pragma unroll
            for (int i = 0; i < 8; ++i) S[(mbase + i) * SLD + colq] = s[i];
        }
    }
    __syncthreads();

    // ---- row softmax; store normalized P to LDS and to global attn (NT) ----
    {
        size_t attnbase = (size_t)Bn * Ltot * Cc
                        + (((size_t)(win * HEADS + h)) * Twin + (size_t)rb * BM) * Twin;
        for (int rr = 0; rr < 8; ++rr) {
            int row = wv * 8 + rr;
            float vals[16];
            float mx = -3.4e38f;
            #pragma unroll
            for (int j = 0; j < 16; ++j) {
                vals[j] = S[row * SLD + lane + 32 * j];
                mx = fmaxf(mx, vals[j]);
            }
            #pragma unroll
            for (int o = 16; o >= 1; o >>= 1) mx = fmaxf(mx, __shfl_xor(mx, o, 32));
            float sum = 0.f;
            #pragma unroll
            for (int j = 0; j < 16; ++j) { vals[j] = __expf(vals[j] - mx); sum += vals[j]; }
            #pragma unroll
            for (int o = 16; o >= 1; o >>= 1) sum += __shfl_xor(sum, o, 32);
            float inv = 1.0f / sum;
            float* srow = S + row * SLD;
            float* arow = out + attnbase + (size_t)row * Twin;
            #pragma unroll
            for (int j = 0; j < 16; ++j) {
                float p = vals[j] * inv;
                srow[lane + 32 * j] = p;                          // for PV
                __builtin_nontemporal_store(p, arow + lane + 32 * j); // attn (NT)
            }
        }
    }
    __syncthreads();

    // ---- O = P V : 4x2 output tiles, one per wave, K = 512 in 16 chunks ----
    {
        int mt = wv >> 1;
        int vt = wv & 1;
        v8f c = {};
        int mrow  = mt * 16 + (lane & 15);
        int koffA = (lane >> 4) * 8;
        int ncol  = vt * 16 + (lane & 15);
        int koffB = (lane >> 4) * 8;
        for (int kt = 0; kt < 16; ++kt) {
            const float* srow = S + mrow * SLD + kt * 32;
            v16h a;
            #pragma unroll
            for (int j = 0; j < 8; ++j) {
                a[j]     = (_Float16)srow[koffA + j];
                a[j + 8] = (_Float16)srow[16 + koffA + j];
            }
            const _Float16* vrow = Vt + ncol * VLD + kt * 32 + koffB;
            v8h blo = *(const v8h*)(vrow);
            v8h bhi = *(const v8h*)(vrow + 16);
            v16h bm;
            #pragma unroll
            for (int i = 0; i < 8; ++i) { bm[i] = blo[i]; bm[8 + i] = bhi[i]; }
            c = __builtin_amdgcn_wmma_f32_16x16x32_f16(
                    false, a, false, bm, (short)0, c, false, false);
        }
        int mbase    = mt * 16 + ((lane >> 4) << 3);
        int chanbase = h * HD + vt * 16 + (lane & 15);
        #pragma unroll
        for (int i = 0; i < 8; ++i) {
            int t = rb * BM + mbase + i;
            int l = (t >> 3) * HWdim + xw * W_SP + (t & 7);
            size_t oidx = ((size_t)b * Ltot + l) * Cc + chanbase;
            out[oidx] = c[i] + out[oidx];    // add LePE written by kernel 1
        }
    }
}

extern "C" void kernel_launch(void* const* d_in, const int* in_sizes, int n_in,
                              void* d_out, int out_size, void* d_ws, size_t ws_size,
                              hipStream_t stream) {
    (void)in_sizes; (void)n_in; (void)out_size; (void)d_ws; (void)ws_size;
    const float* qkv = (const float*)d_in[0];
    const float* lw  = (const float*)d_in[1];
    const float* lb  = (const float*)d_in[2];
    float* out = (float*)d_out;

    // 1) LePE conv into out region
    lepe_kernel<<<(Bn * Ltot * Cc) / 256, 256, 0, stream>>>(qkv, lw, lb, out);

    // 2) windowed attention (adds onto LePE, writes attn slab)
    dim3 grid(Twin / BM, HEADS, NWIN);          // (8, 8, 64)
    attn_kernel<<<grid, 256, SMEM_BYTES, stream>>>(qkv, out);
}
